// AdvancedGCN_45801531244896
// MI455X (gfx1250) — matrix operand order repaired and checked by reading, not compiled, not run on hardware
//
#include <hip/hip_runtime.h>
#include <hip/hip_bf16.h>
#include <math.h>

typedef __attribute__((ext_vector_type(16))) _Float16 v16h;
typedef __attribute__((ext_vector_type(8)))  float    v8f;

#define FLT_MAX_F 3.402823466e+38f
#define CDIV(a,b) (((a)+(b)-1)/(b))

// ---------------- atomic float max via sign-split integer atomics ----------
__device__ __forceinline__ void atomicMaxF(float* addr, float val) {
    if (val >= 0.0f) atomicMax((int*)addr, __float_as_int(val));
    else             atomicMin((unsigned int*)addr, __float_as_uint(val));
}

// ---------------- WMMA fragment builders (16x16x32 f16, contiguous k = 16*half + e in A and B) ------
// A 16x32 f16: lane half (lane>=16) offsets K by 16; VGPR v -> K=2v
__device__ __forceinline__ v16h frag_a(const _Float16* p /* = row*lda + k0 + half*16 */) {
    v16h a;
#pragma unroll
    for (int v = 0; v < 8; ++v) {
        int kb = 2 * v;
        a[2 * v]     = p[kb];
        a[2 * v + 1] = p[kb + 1];
    }
    return a;
}
// B 32x16 f16, read from N-major (transposed) weights: lane half offsets K by 16,
// element e = K offset e -> 16 contiguous f16 per lane.
__device__ __forceinline__ v16h frag_b(const _Float16* p /* = col*ldb + k0 + half*16 */) {
    v16h b;
#pragma unroll
    for (int e = 0; e < 16; ++e) b[e] = p[e];
    return b;
}

// ---------------- generic node GEMM: C[M x 128] = act(A[M x 128] @ W[128 x 128] + b)
__global__ void __launch_bounds__(256)
k_gemm128(const float* __restrict__ A, const float* __restrict__ W,
          const float* __restrict__ bias, float* __restrict__ C, int M, int act) {
    __shared__ _Float16 sA[128 * 128];   // 32 KB, row-major
    __shared__ _Float16 sBt[128 * 128];  // 32 KB, N-major (Wt[n][k])
    const int t = threadIdx.x, lane = t & 31, wave = t >> 5;
    const int row0 = blockIdx.x * 128;

    for (int i = t; i < 128 * 128; i += 256) {          // W (K x N) -> Wt
        int k = i >> 7, n = i & 127;
        sBt[n * 128 + k] = (_Float16)W[i];
        (void)k;
    }
    for (int i = t; i < 128 * 128; i += 256) {          // A tile, clamp tail rows
        int r = i >> 7, c = i & 127;
        int gr = row0 + r; if (gr >= M) gr = M - 1;
        sA[i] = (_Float16)A[(size_t)gr * 128 + c];
    }
    __syncthreads();

    const int half = lane >> 4, lr = lane & 15;
    v8f acc[8] = {};
#pragma unroll
    for (int kt = 0; kt < 4; ++kt) {
        const int k0 = kt * 32;
        v16h af = frag_a(&sA[(wave * 16 + lr) * 128 + k0 + half * 16]);
#pragma unroll
        for (int ct = 0; ct < 8; ++ct) {
            v16h bf = frag_b(&sBt[(ct * 16 + lr) * 128 + k0 + half * 16]);
            acc[ct] = __builtin_amdgcn_wmma_f32_16x16x32_f16(
                false, af, false, bf, (short)0, acc[ct], false, false);
        }
    }
#pragma unroll
    for (int ct = 0; ct < 8; ++ct) {
        const int col = ct * 16 + lr;
        const float bv = bias ? bias[col] : 0.0f;
#pragma unroll
        for (int r = 0; r < 8; ++r) {
            int grow = row0 + wave * 16 + r + 8 * half;
            if (grow < M) {
                float v = acc[ct][r] + bv;
                if (act) v = fmaxf(v, 0.0f);
                C[(size_t)grow * 128 + col] = v;
            }
        }
    }
}

// ---------------- fused EdgeConv: per-edge [h_d, h_s - h_d] -> 256->128 relu ->128,
// atomic segment-max into agg[dst]. 64 edges per block.
__global__ void __launch_bounds__(256)
k_edgeconv(const float* __restrict__ h, const int* __restrict__ src, const int* __restrict__ dst,
           const float* __restrict__ W1, const float* __restrict__ b1,
           const float* __restrict__ W2, const float* __restrict__ b2,
           float* __restrict__ agg, int E) {
    __shared__ _Float16 sW1t[128 * 256]; // 64 KB  W1t[n][k]
    __shared__ _Float16 sW2t[128 * 128]; // 32 KB
    __shared__ _Float16 sA[64 * 256];    // 32 KB  edge message tile
    __shared__ _Float16 sH[64 * 128];    // 16 KB  hidden tile
    const int t = threadIdx.x, lane = t & 31, wave = t >> 5;
    const int e0 = blockIdx.x * 64;

    for (int i = t; i < 256 * 128; i += 256) { int k = i >> 7, n = i & 127; sW1t[n * 256 + k] = (_Float16)W1[i]; }
    for (int i = t; i < 128 * 128; i += 256) { int k = i >> 7, n = i & 127; sW2t[n * 128 + k] = (_Float16)W2[i]; }
    for (int i = t; i < 64 * 256; i += 256) {
        int r = i >> 8, c = i & 255;
        int e = e0 + r; if (e >= E) e = E - 1;
        int s = src[e], d = dst[e];
        float v = (c < 128) ? h[(size_t)d * 128 + c]
                            : (h[(size_t)s * 128 + (c - 128)] - h[(size_t)d * 128 + (c - 128)]);
        sA[i] = (_Float16)v;
    }
    __syncthreads();

    const int rt = wave & 3, cg = wave >> 2;     // 4 row-tiles x 2 col-groups(64 cols)
    const int half = lane >> 4, lr = lane & 15;

    v8f acc[4] = {};
#pragma unroll
    for (int kt = 0; kt < 8; ++kt) {             // K = 256
        const int k0 = kt * 32;
        v16h af = frag_a(&sA[(rt * 16 + lr) * 256 + k0 + half * 16]);
#pragma unroll
        for (int ct = 0; ct < 4; ++ct) {
            v16h bf = frag_b(&sW1t[(cg * 64 + ct * 16 + lr) * 256 + k0 + half * 16]);
            acc[ct] = __builtin_amdgcn_wmma_f32_16x16x32_f16(
                false, af, false, bf, (short)0, acc[ct], false, false);
        }
    }
#pragma unroll
    for (int ct = 0; ct < 4; ++ct) {             // relu(+b1) -> f16 hidden tile
        const int col = cg * 64 + ct * 16 + lr;
        const float bv = b1[col];
#pragma unroll
        for (int r = 0; r < 8; ++r) {
            int row = rt * 16 + r + 8 * half;
            sH[row * 128 + col] = (_Float16)fmaxf(acc[ct][r] + bv, 0.0f);
        }
    }
    __syncthreads();

    v8f acc2[4] = {};
#pragma unroll
    for (int kt = 0; kt < 4; ++kt) {             // K = 128
        const int k0 = kt * 32;
        v16h af = frag_a(&sH[(rt * 16 + lr) * 128 + k0 + half * 16]);
#pragma unroll
        for (int ct = 0; ct < 4; ++ct) {
            v16h bf = frag_b(&sW2t[(cg * 64 + ct * 16 + lr) * 128 + k0 + half * 16]);
            acc2[ct] = __builtin_amdgcn_wmma_f32_16x16x32_f16(
                false, af, false, bf, (short)0, acc2[ct], false, false);
        }
    }
#pragma unroll
    for (int ct = 0; ct < 4; ++ct) {             // segment-max scatter
        const int col = cg * 64 + ct * 16 + lr;
        const float bv = b2[col];
#pragma unroll
        for (int r = 0; r < 8; ++r) {
            int row = rt * 16 + r + 8 * half;
            int e = e0 + row;
            if (e < E) {
                int d = dst[e];
                atomicMaxF(&agg[(size_t)d * 128 + col], acc2[ct][r] + bv);
            }
        }
    }
}

// ---------------- small elementwise / scatter kernels ----------------------
__global__ void k_fill(float* p, float v, int n) {
    int i = blockIdx.x * blockDim.x + threadIdx.x;
    if (i < n) p[i] = v;
}
__global__ void k_deg(const int* __restrict__ dst, float* __restrict__ deg, int E) {
    int e = blockIdx.x * blockDim.x + threadIdx.x;
    if (e < E) atomicAdd(&deg[dst[e]], 1.0f);
}
__global__ void k_rsqrt(float* d, int N) {
    int i = blockIdx.x * blockDim.x + threadIdx.x;
    if (i < N) d[i] = rsqrtf(d[i]);
}
__global__ void k_gcn_scatter(const int* __restrict__ src, const int* __restrict__ dst,
                              const float* __restrict__ dinv, const float* __restrict__ hw,
                              float* __restrict__ out, int E) {
    int idx = blockIdx.x * blockDim.x + threadIdx.x;
    if (idx >= E * 32) return;
    int e = idx >> 5, f = (idx & 31) << 2;
    int s = src[e], d = dst[e];
    float norm = dinv[s] * dinv[d];
    float4 v = *(const float4*)(hw + (size_t)s * 128 + f);
    float* o = out + (size_t)d * 128 + f;
    atomicAdd(o + 0, v.x * norm); atomicAdd(o + 1, v.y * norm);
    atomicAdd(o + 2, v.z * norm); atomicAdd(o + 3, v.w * norm);
}
__global__ void k_gcn_final(const float* __restrict__ accum, const float* __restrict__ hw,
                            const float* __restrict__ dinv, const float* __restrict__ b,
                            float* __restrict__ out, int N) {
    int i = blockIdx.x * blockDim.x + threadIdx.x;
    if (i >= N * 128) return;
    int n = i >> 7, f = i & 127;
    float di = dinv[n];
    out[i] = fmaxf(accum[i] + hw[i] * di * di + b[f], 0.0f);  // self-loop + bias + relu
}
__global__ void k_att(const float* __restrict__ g, const float* __restrict__ as,
                      const float* __restrict__ ad, float* __restrict__ asrc,
                      float* __restrict__ adst, int N) {
    int idx = blockIdx.x * blockDim.x + threadIdx.x;
    if (idx >= N * 4) return;
    int n = idx >> 2, hd = idx & 3;
    float ss = 0.f, sd = 0.f;
    for (int c = 0; c < 32; ++c) {
        float gv = g[(size_t)n * 128 + hd * 32 + c];
        ss += gv * as[hd * 32 + c];
        sd += gv * ad[hd * 32 + c];
    }
    asrc[idx] = ss; adst[idx] = sd;
}
__device__ __forceinline__ float leaky(float x) { return x > 0.f ? x : 0.2f * x; }
__global__ void k_gat_max(const int* __restrict__ src, const int* __restrict__ dst,
                          const float* __restrict__ asrc, const float* __restrict__ adst,
                          float* __restrict__ emax, int E, int N) {
    int idx = blockIdx.x * blockDim.x + threadIdx.x;
    if (idx >= (E + N) * 4) return;
    int i = idx >> 2, hd = idx & 3;
    int s = (i < E) ? src[i] : (i - E);
    int d = (i < E) ? dst[i] : (i - E);
    atomicMaxF(&emax[d * 4 + hd], leaky(asrc[s * 4 + hd] + adst[d * 4 + hd]));
}
__global__ void k_gat_sum(const int* __restrict__ src, const int* __restrict__ dst,
                          const float* __restrict__ asrc, const float* __restrict__ adst,
                          const float* __restrict__ emax, float* __restrict__ esum, int E, int N) {
    int idx = blockIdx.x * blockDim.x + threadIdx.x;
    if (idx >= (E + N) * 4) return;
    int i = idx >> 2, hd = idx & 3;
    int s = (i < E) ? src[i] : (i - E);
    int d = (i < E) ? dst[i] : (i - E);
    float e = leaky(asrc[s * 4 + hd] + adst[d * 4 + hd]);
    atomicAdd(&esum[d * 4 + hd], expf(e - emax[d * 4 + hd]));
}
__global__ void k_gat_scatter(const int* __restrict__ src, const int* __restrict__ dst,
                              const float* __restrict__ asrc, const float* __restrict__ adst,
                              const float* __restrict__ emax, const float* __restrict__ esum,
                              const float* __restrict__ g, float* __restrict__ out, int E, int N) {
    int idx = blockIdx.x * blockDim.x + threadIdx.x;
    if (idx >= (E + N) * 32) return;
    int i = idx >> 5, f = (idx & 31) << 2, hd = f >> 5;
    int s = (i < E) ? src[i] : (i - E);
    int d = (i < E) ? dst[i] : (i - E);
    float e = leaky(asrc[s * 4 + hd] + adst[d * 4 + hd]);
    float alpha = expf(e - emax[d * 4 + hd]) / (esum[d * 4 + hd] + 1e-16f);
    float4 v = *(const float4*)(g + (size_t)s * 128 + f);
    float* o = out + (size_t)d * 128 + f;
    atomicAdd(o + 0, v.x * alpha); atomicAdd(o + 1, v.y * alpha);
    atomicAdd(o + 2, v.z * alpha); atomicAdd(o + 3, v.w * alpha);
}
__global__ void k_elu_bias(const float* __restrict__ in, const float* __restrict__ b,
                           float* __restrict__ out, int N) {
    int i = blockIdx.x * blockDim.x + threadIdx.x;
    if (i >= N * 128) return;
    float v = in[i] + b[i & 127];
    out[i] = v > 0.f ? v : (expf(v) - 1.0f);
}
__global__ void k_relu_floor(const float* __restrict__ in, float* __restrict__ out, int n) {
    int i = blockIdx.x * blockDim.x + threadIdx.x;
    if (i < n) out[i] = fmaxf(in[i], 0.0f);   // -FLT_MAX (no in-edge) -> 0, relu otherwise
}
__global__ void k_gin_scatter(const int* __restrict__ src, const int* __restrict__ dst,
                              const float* __restrict__ h, float* __restrict__ out, int E) {
    int idx = blockIdx.x * blockDim.x + threadIdx.x;
    if (idx >= E * 32) return;
    int e = idx >> 5, f = (idx & 31) << 2;
    int s = src[e], d = dst[e];
    float4 v = *(const float4*)(h + (size_t)s * 128 + f);
    float* o = out + (size_t)d * 128 + f;
    atomicAdd(o + 0, v.x); atomicAdd(o + 1, v.y);
    atomicAdd(o + 2, v.z); atomicAdd(o + 3, v.w);
}
__global__ void k_add(const float* a, const float* b, float* out, int n) {
    int i = blockIdx.x * blockDim.x + threadIdx.x;
    if (i < n) out[i] = a[i] + b[i];
}
__global__ void k_gate2(const float* __restrict__ hC, const float* __restrict__ W2,
                        const float* __restrict__ b2, float* __restrict__ gate, int N) {
    int n = blockIdx.x * blockDim.x + threadIdx.x;
    if (n >= N) return;
    float s = b2[0];
    for (int k = 0; k < 128; ++k) s += hC[(size_t)n * 128 + k] * W2[k];
    gate[n] = s;
}
__global__ void k_gmax(const int* __restrict__ batch, const float* __restrict__ gate,
                       float* __restrict__ gmax, int N) {
    int n = blockIdx.x * blockDim.x + threadIdx.x;
    if (n < N) atomicMaxF(&gmax[batch[n]], gate[n]);
}
__global__ void k_gsum(const int* __restrict__ batch, const float* __restrict__ gate,
                       const float* __restrict__ gmax, float* __restrict__ gsum, int N) {
    int n = blockIdx.x * blockDim.x + threadIdx.x;
    if (n < N) { int b = batch[n]; atomicAdd(&gsum[b], expf(gate[n] - gmax[b])); }
}
__global__ void k_pool(const int* __restrict__ batch, const float* __restrict__ gate,
                       const float* __restrict__ gmax, const float* __restrict__ gsum,
                       const float* __restrict__ h, float* __restrict__ pooled, int N) {
    int idx = blockIdx.x * blockDim.x + threadIdx.x;
    if (idx >= N * 32) return;
    int n = idx >> 5, f = (idx & 31) << 2;
    int b = batch[n];
    float w = expf(gate[n] - gmax[b]) / (gsum[b] + 1e-16f);
    float4 v = *(const float4*)(h + (size_t)n * 128 + f);
    float* o = pooled + (size_t)b * 128 + f;
    atomicAdd(o + 0, v.x * w); atomicAdd(o + 1, v.y * w);
    atomicAdd(o + 2, v.z * w); atomicAdd(o + 3, v.w * w);
}
__global__ void k_head(const float* __restrict__ pooled, const float* __restrict__ fcW,
                       const float* __restrict__ fcb, float* __restrict__ out, int G) {
    __shared__ float lg[1024];
    int idx = threadIdx.x;
    int total = G * 10;
    if (idx < total) {
        int g = idx / 10, o = idx % 10;
        float s = fcb[o];
        for (int k = 0; k < 128; ++k) s += pooled[(size_t)g * 128 + k] * fcW[k * 10 + o];
        lg[idx] = s;
    }
    __syncthreads();
    if (idx < total) {
        int g = idx / 10;
        float m = -FLT_MAX_F;
        for (int o = 0; o < 10; ++o) m = fmaxf(m, lg[g * 10 + o]);
        float se = 0.f;
        for (int o = 0; o < 10; ++o) se += expf(lg[g * 10 + o] - m);
        out[idx] = lg[idx] - m - logf(se);
    }
}

// ---------------- driver ---------------------------------------------------
extern "C" void kernel_launch(void* const* d_in, const int* in_sizes, int n_in,
                              void* d_out, int out_size, void* d_ws, size_t ws_size,
                              hipStream_t stream) {
    (void)n_in; (void)ws_size;
    const float* x       = (const float*)d_in[0];
    const int*   ei      = (const int*)  d_in[1];
    const int*   batch   = (const int*)  d_in[2];
    const float* gcn_W   = (const float*)d_in[3];
    const float* gcn_b   = (const float*)d_in[4];
    const float* gat_W   = (const float*)d_in[5];
    const float* att_src = (const float*)d_in[6];
    const float* att_dst = (const float*)d_in[7];
    const float* gat_b   = (const float*)d_in[8];
    const float* ec_W1   = (const float*)d_in[9];
    const float* ec_b1   = (const float*)d_in[10];
    const float* ec_W2   = (const float*)d_in[11];
    const float* ec_b2   = (const float*)d_in[12];
    const float* gin_W1  = (const float*)d_in[13];
    const float* gin_b1  = (const float*)d_in[14];
    const float* gin_W2  = (const float*)d_in[15];
    const float* gin_b2  = (const float*)d_in[16];
    const float* gate_W1 = (const float*)d_in[17];
    const float* gate_b1 = (const float*)d_in[18];
    const float* gate_W2 = (const float*)d_in[19];
    const float* gate_b2 = (const float*)d_in[20];
    const float* fc_W    = (const float*)d_in[21];
    const float* fc_b    = (const float*)d_in[22];

    const int N = in_sizes[0] / 128;
    const int E = in_sizes[1] / 2;
    const int G = out_size / 10;
    const int* src = ei;
    const int* dst = ei + E;

    float* ws     = (float*)d_ws;
    float* hA     = ws;                         // N*128
    float* hB     = hA + (size_t)N * 128;       // N*128
    float* hC     = hB + (size_t)N * 128;       // N*128
    float* dinv   = hC + (size_t)N * 128;       // N
    float* asrc   = dinv + N;                   // N*4
    float* adst   = asrc + (size_t)N * 4;       // N*4
    float* emax   = adst + (size_t)N * 4;       // N*4
    float* esum   = emax + (size_t)N * 4;       // N*4
    float* gatev  = esum + (size_t)N * 4;       // N
    float* gmax   = gatev + N;                  // G
    float* gsum   = gmax + G;                   // G
    float* pooled = gsum + G;                   // G*128

    // ---- GCN ----
    k_fill<<<CDIV(N, 256), 256, 0, stream>>>(dinv, 1.0f, N);            // deg incl. self loop
    k_deg<<<CDIV(E, 256), 256, 0, stream>>>(dst, dinv, E);
    k_rsqrt<<<CDIV(N, 256), 256, 0, stream>>>(dinv, N);
    k_gemm128<<<CDIV(N, 128), 256, 0, stream>>>(x, gcn_W, nullptr, hC, N, 0);  // XW
    k_fill<<<CDIV(N * 128, 256), 256, 0, stream>>>(hB, 0.0f, N * 128);
    k_gcn_scatter<<<CDIV(E * 32, 256), 256, 0, stream>>>(src, dst, dinv, hC, hB, E);
    k_gcn_final<<<CDIV(N * 128, 256), 256, 0, stream>>>(hB, hC, dinv, gcn_b, hA, N);

    // ---- GAT ----
    k_gemm128<<<CDIV(N, 128), 256, 0, stream>>>(hA, gat_W, nullptr, hC, N, 0); // g
    k_att<<<CDIV(N * 4, 256), 256, 0, stream>>>(hC, att_src, att_dst, asrc, adst, N);
    k_fill<<<CDIV(N * 4, 256), 256, 0, stream>>>(emax, -FLT_MAX_F, N * 4);
    k_fill<<<CDIV(N * 4, 256), 256, 0, stream>>>(esum, 0.0f, N * 4);
    k_gat_max<<<CDIV((E + N) * 4, 256), 256, 0, stream>>>(src, dst, asrc, adst, emax, E, N);
    k_gat_sum<<<CDIV((E + N) * 4, 256), 256, 0, stream>>>(src, dst, asrc, adst, emax, esum, E, N);
    k_fill<<<CDIV(N * 128, 256), 256, 0, stream>>>(hB, 0.0f, N * 128);
    k_gat_scatter<<<CDIV((E + N) * 32, 256), 256, 0, stream>>>(src, dst, asrc, adst, emax, esum, hC, hB, E, N);
    k_elu_bias<<<CDIV(N * 128, 256), 256, 0, stream>>>(hB, gat_b, hA, N);

    // ---- EdgeConv (fused WMMA MLP + segment max) ----
    k_fill<<<CDIV(N * 128, 256), 256, 0, stream>>>(hB, -FLT_MAX_F, N * 128);
    k_edgeconv<<<CDIV(E, 64), 256, 0, stream>>>(hA, src, dst, ec_W1, ec_b1, ec_W2, ec_b2, hB, E);
    k_relu_floor<<<CDIV(N * 128, 256), 256, 0, stream>>>(hB, hA, N * 128);

    // ---- GIN ----
    k_fill<<<CDIV(N * 128, 256), 256, 0, stream>>>(hC, 0.0f, N * 128);
    k_gin_scatter<<<CDIV(E * 32, 256), 256, 0, stream>>>(src, dst, hA, hC, E);
    k_add<<<CDIV(N * 128, 256), 256, 0, stream>>>(hA, hC, hB, N * 128);        // s = h + sum
    k_gemm128<<<CDIV(N, 128), 256, 0, stream>>>(hB, gin_W1, gin_b1, hC, N, 1);
    k_gemm128<<<CDIV(N, 128), 256, 0, stream>>>(hC, gin_W2, gin_b2, hB, N, 1); // final h in hB

    // ---- GlobalAttention pooling ----
    k_gemm128<<<CDIV(N, 128), 256, 0, stream>>>(hB, gate_W1, gate_b1, hC, N, 1);
    k_gate2<<<CDIV(N, 256), 256, 0, stream>>>(hC, gate_W2, gate_b2, gatev, N);
    k_fill<<<CDIV(G, 64), 64, 0, stream>>>(gmax, -FLT_MAX_F, G);
    k_fill<<<CDIV(G, 64), 64, 0, stream>>>(gsum, 0.0f, G);
    k_fill<<<CDIV(G * 128, 256), 256, 0, stream>>>(pooled, 0.0f, G * 128);
    k_gmax<<<CDIV(N, 256), 256, 0, stream>>>(batch, gatev, gmax, N);
    k_gsum<<<CDIV(N, 256), 256, 0, stream>>>(batch, gatev, gmax, gsum, N);
    k_pool<<<CDIV(N * 32, 256), 256, 0, stream>>>(batch, gatev, gmax, gsum, hB, pooled, N);

    // ---- head + log_softmax ----
    int th = CDIV(G * 10, 64) * 64; if (th > 1024) th = 1024;
    k_head<<<1, th, 0, stream>>>(pooled, fc_W, fc_b, (float*)d_out, G);
}
